// MaxRetrievalModel_80169859547697
// MI455X (gfx1250) — matrix operand order repaired and checked by reading, not compiled, not run on hardware
//
#include <hip/hip_runtime.h>
#include <hip/hip_bf16.h>
#include <math.h>

// ---------------------------------------------------------------------------
// Problem constants (from reference)
// ---------------------------------------------------------------------------
#define BB 64
#define NN 8192
#define DITEM 64
#define DEMB 256
#define NCLS 1000
#define TM 64                 // item rows per block in the hot kernel
#define NTILES (NN / TM)      // 128

typedef __attribute__((ext_vector_type(16))) _Float16 v16h;
typedef __attribute__((ext_vector_type(8)))  _Float16 v8h;
typedef __attribute__((ext_vector_type(8)))  float    v8f;

union Frag16 { v16h v; v8h h2[2]; _Float16 e[16]; };

// Exact erf-based GELU (tiny kernels: query / head)
__device__ __forceinline__ float gelu_exact(float x) {
    return 0.5f * x * (1.0f + erff(x * 0.70710678118654752f));
}

// Fast erf (Abramowitz-Stegun 7.1.26, |eps| <= 1.5e-7): one v_rcp, one v_exp,
// ~6 FMAs.  Used in the hot per-item kernel where erff's libm expansion would
// rival the WMMA time on the VALU pipe.
__device__ __forceinline__ float fast_erf(float x) {
    float ax = fabsf(x);
    float t  = __builtin_amdgcn_rcpf(1.0f + 0.3275911f * ax);
    float p  = t * (0.254829592f +
               t * (-0.284496736f +
               t * (1.421413741f +
               t * (-1.453152027f +
               t * 1.061405429f))));
    float r  = 1.0f - p * __expf(-ax * ax);
    return copysignf(r, x);
}
__device__ __forceinline__ float gelu_fast(float x) {
    return 0.5f * x * (1.0f + fast_erf(x * 0.70710678118654752f));
}

// ---------------------------------------------------------------------------
// Kernel P: repack psi_x_w1 [64,256] and psi_x_w2 [256,256] (fp32, row-major
// [in,out]) into f16 WMMA B-fragment order for v_wmma_f32_16x16x32_f16.
// B (32x16, f16): lane L holds column n = L%16; element i of v16h holds
// K = (L/16)*16 + i (this code's contiguous k order; A loads match).  Packed as [frag = kk*16+ct][lane][i] contiguous.
// ---------------------------------------------------------------------------
__global__ __launch_bounds__(256) void k_pack(const float* __restrict__ w1,
                                              const float* __restrict__ w2,
                                              _Float16* __restrict__ pW1,
                                              _Float16* __restrict__ pW2) {
    int e = blockIdx.x * 256 + threadIdx.x;
    const int W1E = 2 * 16 * 512;   // 16384
    const int W2E = 8 * 16 * 512;   // 65536
    if (e < W1E) {
        int frag = e >> 9, lane = (e >> 4) & 31, hi = e & 15;
        int kk = frag >> 4, ct = frag & 15;
        int k = kk * 32 + (lane >> 4) * 16 + hi;
        int n = ct * 16 + (lane & 15);
        pW1[e] = (_Float16)w1[k * DEMB + n];
    } else if (e < W1E + W2E) {
        int i = e - W1E;
        int frag = i >> 9, lane = (i >> 4) & 31, hi = i & 15;
        int kk = frag >> 4, ct = frag & 15;
        int k = kk * 32 + (lane >> 4) * 16 + hi;
        int n = ct * 16 + (lane & 15);
        pW2[i] = (_Float16)w2[k * DEMB + n];
    }
}

// ---------------------------------------------------------------------------
// Kernel Q: per-batch query path.  q = (GELU(xq*w1+b1)@w2+b2)@q_w+q_b, then
// kq = k_w q (length-256) and qkb = q . k_b.   One block per batch.
// ---------------------------------------------------------------------------
__global__ __launch_bounds__(256) void k_query(
    const float* __restrict__ x_query,
    const float* __restrict__ pqw1, const float* __restrict__ pqb1,
    const float* __restrict__ pqw2, const float* __restrict__ pqb2,
    const float* __restrict__ q_w,  const float* __restrict__ q_b,
    const float* __restrict__ k_w,  const float* __restrict__ k_b,
    float* __restrict__ kq, float* __restrict__ qkb) {
    __shared__ float tq[DEMB], hq[DEMB], qv[DEMB], red[256];
    int b = blockIdx.x, j = threadIdx.x;
    float xq = x_query[b];
    tq[j] = gelu_exact(xq * pqw1[j] + pqb1[j]);
    __syncthreads();
    float a = pqb2[j];
    for (int i = 0; i < DEMB; ++i) a += tq[i] * pqw2[i * DEMB + j];
    hq[j] = a;
    __syncthreads();
    float qj = q_b[j];
    for (int i = 0; i < DEMB; ++i) qj += hq[i] * q_w[i * DEMB + j];
    qv[j] = qj;
    __syncthreads();
    float kqj = 0.0f;
    for (int d = 0; d < DEMB; ++d) kqj += k_w[j * DEMB + d] * qv[d];
    kq[b * DEMB + j] = kqj;
    red[j] = qv[j] * k_b[j];
    __syncthreads();
    for (int s = 128; s > 0; s >>= 1) {
        if (j < s) red[j] += red[j + s];
        __syncthreads();
    }
    if (j == 0) qkb[b] = red[0];
}

// ---------------------------------------------------------------------------
// Kernel I (hot): per 64-item tile: h = GELU(x@W1+b1)@W2+b2 via WMMA f16,
// h -> f16 scratch; scores = h . kq + qkb -> scratch.
// Block = 256 threads = 8 waves; wave w handles row tile (w>>1)*16 and
// column tiles ct in [ (w&1)*8 , +8 ).
// x tile is staged with CDNA5 async global->LDS DMA (ASYNCcnt).
// ---------------------------------------------------------------------------
struct SmemItems {
    union {
        float    xraw[TM][DITEM];    // 16 KB f32 staging (dead after convert)
        _Float16 hbuf[TM][DEMB + 8]; // h1 then h (stride 264 -> 528B rows)
    } u;
    _Float16 xs[TM][DITEM + 8];      // x tile in f16 (stride 72 -> 144B rows)
    float kq[DEMB];
    float sc[TM][4];
    float qkb;
};

__global__ __launch_bounds__(256) void k_items(
    const float* __restrict__ x_items,
    const _Float16* __restrict__ pW1, const float* __restrict__ b1,
    const _Float16* __restrict__ pW2, const float* __restrict__ b2,
    const float* __restrict__ kq, const float* __restrict__ qkb,
    _Float16* __restrict__ hout, float* __restrict__ scores) {
    __shared__ SmemItems s;
    const int tid = threadIdx.x;
    const int lane = tid & 31;
    const int w = tid >> 5;
    const int m0 = (w >> 1) * 16;
    const int cb = (w & 1) * 8;
    const int b = blockIdx.y;
    const int n0 = blockIdx.x * TM;
    const int hh = lane >> 4;
    const int mrow = m0 + (lane & 15);

    s.kq[tid] = kq[b * DEMB + tid];
    if (tid == 0) s.qkb = qkb[b];

    // ---- stage x tile via async global->LDS (b128 per lane, ASYNCcnt) ----
    const float* xg = x_items + ((size_t)b * NN + n0) * DITEM;
    {
        unsigned lds_base = (unsigned)(uintptr_t)&s.u.xraw[0][0];
        #pragma unroll
        for (int i = 0; i < 4; ++i) {
            unsigned chunk = (unsigned)tid + (unsigned)i * 256u; // of 1024
            unsigned byte_off = chunk * 16u;                     // 16B chunks
            unsigned lds_addr = lds_base + byte_off;
            asm volatile("global_load_async_to_lds_b128 %0, %1, %2"
                         :
                         : "v"(lds_addr), "v"(byte_off), "s"(xg)
                         : "memory");
        }
        asm volatile("s_wait_asynccnt 0" ::: "memory");
    }
    __syncthreads();
    // convert staged f32 tile -> f16 xs
    #pragma unroll
    for (int i = 0; i < 16; ++i) {
        int idx = tid + i * 256;
        int r = idx >> 6, c = idx & 63;
        s.xs[r][c] = (_Float16)s.u.xraw[r][c];
    }
    __syncthreads();

    // ---- GEMM1: [64,64] @ [64,256], K = 2 x 32 ----
    Frag16 af, bf;
    v8f c1[8];
    #pragma unroll
    for (int j = 0; j < 8; ++j)
        #pragma unroll
        for (int g = 0; g < 8; ++g) c1[j][g] = 0.0f;

    #pragma unroll
    for (int kk = 0; kk < 2; ++kk) {
        int kb = kk * 32 + hh * 16;
        af.h2[0] = *(const v8h*)&s.xs[mrow][kb];
        af.h2[1] = *(const v8h*)&s.xs[mrow][kb + 8];
        #pragma unroll
        for (int j = 0; j < 8; ++j) {
            const v8h* bp = (const v8h*)(pW1 + (((size_t)(kk * 16 + cb + j)) * 32 + lane) * 16);
            bf.h2[0] = bp[0];
            bf.h2[1] = bp[1];
            c1[j] = __builtin_amdgcn_wmma_f32_16x16x32_f16(
                false, af.v, false, bf.v, (short)0, c1[j], false, false);
        }
    }
    // bias + fast-erf GELU -> h1 (f16) into hbuf (xraw is dead now)
    #pragma unroll
    for (int j = 0; j < 8; ++j) {
        int n = (cb + j) * 16 + (lane & 15);
        float bias = b1[n];
        #pragma unroll
        for (int g = 0; g < 8; ++g) {
            int m = m0 + g + 8 * hh;
            s.u.hbuf[m][n] = (_Float16)gelu_fast(c1[j][g] + bias);
        }
    }
    __syncthreads();

    // ---- GEMM2: [64,256] @ [256,256], K = 8 x 32 ----
    v8f c2[8];
    #pragma unroll
    for (int j = 0; j < 8; ++j)
        #pragma unroll
        for (int g = 0; g < 8; ++g) c2[j][g] = 0.0f;

    #pragma unroll
    for (int kk = 0; kk < 8; ++kk) {
        int kb = kk * 32 + hh * 16;
        af.h2[0] = *(const v8h*)&s.u.hbuf[mrow][kb];
        af.h2[1] = *(const v8h*)&s.u.hbuf[mrow][kb + 8];
        #pragma unroll
        for (int j = 0; j < 8; ++j) {
            const v8h* bp = (const v8h*)(pW2 + (((size_t)(kk * 16 + cb + j)) * 32 + lane) * 16);
            bf.h2[0] = bp[0];
            bf.h2[1] = bp[1];
            c2[j] = __builtin_amdgcn_wmma_f32_16x16x32_f16(
                false, af.v, false, bf.v, (short)0, c2[j], false, false);
        }
    }
    __syncthreads();           // everyone done reading h1 from hbuf
    #pragma unroll
    for (int j = 0; j < 8; ++j) {
        int n = (cb + j) * 16 + (lane & 15);
        float bias = b2[n];
        #pragma unroll
        for (int g = 0; g < 8; ++g) {
            int m = m0 + g + 8 * hh;
            s.u.hbuf[m][n] = (_Float16)(c2[j][g] + bias);
        }
    }
    __syncthreads();

    // ---- write h (f16, row-major) to global scratch ----
    _Float16* hg = hout + ((size_t)b * NN + n0) * DEMB;
    #pragma unroll
    for (int i = 0; i < 8; ++i) {
        int chunk = tid + i * 256;          // 2048 chunks of 8 halves
        int r = chunk >> 5;
        int c8 = (chunk & 31) * 8;
        *(v8h*)(hg + (size_t)r * DEMB + c8) = *(const v8h*)&s.u.hbuf[r][c8];
    }

    // ---- scores = h . kq + qkb ----
    {
        int m = tid & 63, part = tid >> 6;
        float acc = 0.0f;
        const float* kqs = s.kq + part * 64;
        const _Float16* hr = &s.u.hbuf[m][part * 64];
        #pragma unroll 8
        for (int d = 0; d < 64; ++d) acc += (float)hr[d] * kqs[d];
        s.sc[m][part] = acc;
    }
    __syncthreads();
    if (tid < 64) {
        float sv = s.sc[tid][0] + s.sc[tid][1] + s.sc[tid][2] + s.sc[tid][3] + s.qkb;
        scores[(size_t)b * NN + n0 + tid] = sv;
    }
}

// ---------------------------------------------------------------------------
// Kernel S: per-batch 32-iteration bisection for lambda (q=1 Stieltjes).
// ---------------------------------------------------------------------------
__global__ __launch_bounds__(256) void k_stieltjes(const float* __restrict__ scores,
                                                   float* __restrict__ lamx) {
    __shared__ float sc[NN];
    __shared__ float red[256];
    __shared__ float bounds[2];
    int b = blockIdx.x, tid = threadIdx.x;
    const float* sg = scores + (size_t)b * NN;
    float mx = -1e30f;
    for (int i = tid; i < NN; i += 256) {
        float v = fminf(fmaxf(sg[i], -50.0f), 50.0f);
        sc[i] = v;
        mx = fmaxf(mx, v);
    }
    red[tid] = mx;
    __syncthreads();
    for (int st = 128; st > 0; st >>= 1) {
        if (tid < st) red[tid] = fmaxf(red[tid], red[tid + st]);
        __syncthreads();
    }
    float xmax = red[0];
    __syncthreads();
    if (tid == 0) { bounds[0] = 1e-9f; bounds[1] = (float)NN; }
    __syncthreads();
    for (int it = 0; it < 32; ++it) {
        float mid = (bounds[0] + bounds[1]) * 0.5f;
        float acc = 0.0f;
        for (int i = tid; i < NN; i += 256) {
            float d = mid - (sc[i] - xmax);
            acc += 1.0f / fmaxf(d, 1e-9f);
        }
        red[tid] = acc;
        __syncthreads();
        for (int st = 128; st > 0; st >>= 1) {
            if (tid < st) red[tid] += red[tid + st];
            __syncthreads();
        }
        if (tid == 0) {
            if (red[0] - 1.0f > 0.0f) bounds[0] = mid; else bounds[1] = mid;
        }
        __syncthreads();
    }
    if (tid == 0) {
        lamx[b * 2 + 0] = (bounds[0] + bounds[1]) * 0.5f;
        lamx[b * 2 + 1] = xmax;
    }
}

// ---------------------------------------------------------------------------
// Kernel A: zacc[b][d] += sum_n attn_n * h[b][n][d] ; sumattn[b] += sum attn.
// Grid (8, B); each block handles a 1024-row slab.  Vectorized: each thread
// owns 8 consecutive columns (v8h loads), 8 row-groups, LDS partial reduce,
// one global atomic per column per block.
// ---------------------------------------------------------------------------
__global__ __launch_bounds__(256) void k_attend(const float* __restrict__ scores,
                                                const float* __restrict__ lamx,
                                                const _Float16* __restrict__ h,
                                                float* __restrict__ zacc,
                                                float* __restrict__ sumattn) {
    const int CH = NN / 8;                 // 1024
    __shared__ float attn[CH];
    __shared__ float red[256];
    __shared__ float zs[8][DEMB];
    int b = blockIdx.y, n0 = blockIdx.x * CH, tid = threadIdx.x;
    float lam = lamx[b * 2 + 0], xmax = lamx[b * 2 + 1];
    float psum = 0.0f;
    for (int i = tid; i < CH; i += 256) {
        float v = fminf(fmaxf(scores[(size_t)b * NN + n0 + i], -50.0f), 50.0f);
        float wgt = 1.0f / fmaxf(lam - (v - xmax), 1e-9f);
        attn[i] = wgt;
        psum += wgt;
    }
    red[tid] = psum;
    __syncthreads();
    for (int st = 128; st > 0; st >>= 1) {
        if (tid < st) red[tid] += red[tid + st];
        __syncthreads();
    }
    if (tid == 0) atomicAdd(&sumattn[b], red[0]);

    const int g = tid >> 5;                // row group 0..7
    const int dbase = (tid & 31) * 8;      // 8 consecutive columns
    float acc[8];
    #pragma unroll
    for (int j = 0; j < 8; ++j) acc[j] = 0.0f;
    const _Float16* hb = h + ((size_t)b * NN + n0) * DEMB + dbase;
    for (int i = g; i < CH; i += 8) {
        v8h hv = *(const v8h*)(hb + (size_t)i * DEMB);
        float wa = attn[i];
        #pragma unroll
        for (int j = 0; j < 8; ++j) acc[j] += wa * (float)hv[j];
    }
    #pragma unroll
    for (int j = 0; j < 8; ++j) zs[g][dbase + j] = acc[j];
    __syncthreads();
    {
        float t = 0.0f;
        #pragma unroll
        for (int g2 = 0; g2 < 8; ++g2) t += zs[g2][tid];
        atomicAdd(&zacc[b * DEMB + tid], t);
    }
}

// ---------------------------------------------------------------------------
// Kernel H: head. z = zacc@v_w + sumattn*v_b ; out = GELU(z@phi1+b1)@phi2+b2.
// ---------------------------------------------------------------------------
__global__ __launch_bounds__(256) void k_head(const float* __restrict__ zacc,
                                              const float* __restrict__ sumattn,
                                              const float* __restrict__ v_w,
                                              const float* __restrict__ v_b,
                                              const float* __restrict__ phi_w1,
                                              const float* __restrict__ phi_b1,
                                              const float* __restrict__ phi_w2,
                                              const float* __restrict__ phi_b2,
                                              float* __restrict__ out) {
    __shared__ float zl[DEMB], t1[DEMB], t2[DEMB];
    int b = blockIdx.x, j = threadIdx.x;
    zl[j] = zacc[b * DEMB + j];
    __syncthreads();
    float sa = sumattn[b];
    float acc = sa * v_b[j];
    for (int e = 0; e < DEMB; ++e) acc += zl[e] * v_w[e * DEMB + j];
    t1[j] = acc;
    __syncthreads();
    acc = phi_b1[j];
    for (int e = 0; e < DEMB; ++e) acc += t1[e] * phi_w1[e * DEMB + j];
    t2[j] = gelu_exact(acc);
    __syncthreads();
    for (int c = j; c < NCLS; c += 256) {
        float o = phi_b2[c];
        for (int e = 0; e < DEMB; ++e) o += t2[e] * phi_w2[e * NCLS + c];
        out[b * NCLS + c] = o;
    }
}

// ---------------------------------------------------------------------------
// Launcher
// ---------------------------------------------------------------------------
extern "C" void kernel_launch(void* const* d_in, const int* in_sizes, int n_in,
                              void* d_out, int out_size, void* d_ws, size_t ws_size,
                              hipStream_t stream) {
    const float* x_items  = (const float*)d_in[0];
    const float* x_query  = (const float*)d_in[1];
    const float* psi_x_w1 = (const float*)d_in[2];
    const float* psi_x_b1 = (const float*)d_in[3];
    const float* psi_x_w2 = (const float*)d_in[4];
    const float* psi_x_b2 = (const float*)d_in[5];
    const float* psi_q_w1 = (const float*)d_in[6];
    const float* psi_q_b1 = (const float*)d_in[7];
    const float* psi_q_w2 = (const float*)d_in[8];
    const float* psi_q_b2 = (const float*)d_in[9];
    const float* q_w      = (const float*)d_in[10];
    const float* q_b      = (const float*)d_in[11];
    const float* k_w      = (const float*)d_in[12];
    const float* k_b      = (const float*)d_in[13];
    const float* v_w      = (const float*)d_in[14];
    const float* v_b      = (const float*)d_in[15];
    const float* phi_w1   = (const float*)d_in[16];
    const float* phi_b1   = (const float*)d_in[17];
    const float* phi_w2   = (const float*)d_in[18];
    const float* phi_b2   = (const float*)d_in[19];
    float* out = (float*)d_out;

    char* ws = (char*)d_ws;
    // scratch layout (bytes)
    _Float16* pW1     = (_Float16*)(ws + 0);          //  32768
    _Float16* pW2     = (_Float16*)(ws + 32768);      // 131072
    float*    kq      = (float*)   (ws + 163840);     //  65536
    float*    qkb     = (float*)   (ws + 229376);     //    256
    float*    lamx    = (float*)   (ws + 229632);     //    512
    float*    zacc    = (float*)   (ws + 230144);     //  65536
    float*    sumattn = (float*)   (ws + 295680);     //    256
    float*    scores  = (float*)   (ws + 295936);     // 2097152
    _Float16* hbuf    = (_Float16*)(ws + 2393088);    // 268435456

    // 1) repack psi_x weights into WMMA B-fragment order (f16)
    k_pack<<<dim3(320), dim3(256), 0, stream>>>(psi_x_w1, psi_x_w2, pW1, pW2);

    // 2) query path -> kq, qkb
    k_query<<<dim3(BB), dim3(256), 0, stream>>>(x_query,
        psi_q_w1, psi_q_b1, psi_q_w2, psi_q_b2, q_w, q_b, k_w, k_b, kq, qkb);

    // zero accumulators (zacc..sumattn contiguous region)
    hipMemsetAsync(ws + 230144, 0, 65792, stream);

    // 3) hot kernel: item MLP via WMMA + scores
    k_items<<<dim3(NTILES, BB), dim3(256), 0, stream>>>(
        x_items, pW1, psi_x_b1, pW2, psi_x_b2, kq, qkb, hbuf, scores);

    // 4) per-batch bisection
    k_stieltjes<<<dim3(BB), dim3(256), 0, stream>>>(scores, lamx);

    // 5) attention-weighted accumulation of h
    k_attend<<<dim3(8, BB), dim3(256), 0, stream>>>(scores, lamx, hbuf, zacc, sumattn);

    // 6) head
    k_head<<<dim3(BB), dim3(256), 0, stream>>>(zacc, sumattn, v_w, v_b,
        phi_w1, phi_b1, phi_w2, phi_b2, out);
}
